// QLoraLinear_4475355922883
// MI455X (gfx1250) — matrix operand order repaired and checked by reading, not compiled, not run on hardware
//
#include <hip/hip_runtime.h>

typedef __attribute__((ext_vector_type(16))) __bf16 bf16x16;
typedef __attribute__((ext_vector_type(8)))  __bf16 bf16x8;
typedef __attribute__((ext_vector_type(4)))  __bf16 bf16x4;
typedef __attribute__((ext_vector_type(8)))  float  f32x8;

#define IN_F   4096
#define OUT_F  4096
#define M_TOT  8192    // 4 * 2048
#define LDA    48      // padded LDS stride for A tile (bf16 elems): 96B rows, 16B-aligned chunks
#define LDB    264     // padded LDS stride for B tile (bf16 elems): 528B rows, 16B-aligned chunks
#define BN     256     // block N tile
#define BM     128     // block M tile

// ---------------------------------------------------------------------------
// Kernel 1: dequantize int4 weights, fold LoRA delta (2.0 * B@A), store the
// result TRANSPOSED as bf16:
//   Wt[k][o] = (q[o][k]-8)*scale[o][k/64] + 2*sum_r loraB[o][r]*loraA[r][k]
// One block handles a 64(o) x 64(i) tile; i-tile aligns with the quant block,
// so each output row uses a single scale. Transpose via LDS for coalesced I/O.
// ---------------------------------------------------------------------------
__global__ __launch_bounds__(256) void dequant_fold_kernel(
    const int* __restrict__ q, const float* __restrict__ scales,
    const float* __restrict__ loraA, const float* __restrict__ loraB,
    __bf16* __restrict__ Wt)
{
    __shared__ __align__(16) float  sA[16][64];  // loraA tile [r][i]
    __shared__ __align__(16) __bf16 sT[64][72];  // transposed weights [i][o], padded

    const int t  = threadIdx.x;
    const int i0 = blockIdx.x * 64;
    const int o0 = blockIdx.y * 64;

    {   // stage loraA tile: 16 rows x 64 cols = 256 float4 -> one per thread
        const int r = t >> 4;
        const int c = (t & 15) * 4;
        const float4 v = *(const float4*)(loraA + (size_t)r * IN_F + i0 + c);
        sA[r][c + 0] = v.x; sA[r][c + 1] = v.y;
        sA[r][c + 2] = v.z; sA[r][c + 3] = v.w;
    }
    __syncthreads();

    const int orow = t >> 2;            // 0..63
    const int iseg = (t & 3) * 16;      // 0,16,32,48
    const int o    = o0 + orow;
    const float scale = scales[(size_t)o * (IN_F / 64) + blockIdx.x];

    float bR[16];
    {
        const float4* bp = (const float4*)(loraB + (size_t)o * 16);
        #pragma unroll
        for (int r4 = 0; r4 < 4; ++r4) {
            float4 b = bp[r4];
            bR[4 * r4 + 0] = b.x; bR[4 * r4 + 1] = b.y;
            bR[4 * r4 + 2] = b.z; bR[4 * r4 + 3] = b.w;
        }
    }

    const int* qp = q + (size_t)o * IN_F + i0 + iseg;
    #pragma unroll
    for (int j = 0; j < 16; j += 4) {
        const int4 qv = *(const int4*)(qp + j);
        const int codes[4] = { qv.x, qv.y, qv.z, qv.w };
        #pragma unroll
        for (int e = 0; e < 4; ++e) {
            const int il = iseg + j + e;
            float dot = 0.0f;
            #pragma unroll
            for (int r = 0; r < 16; ++r)
                dot = __builtin_fmaf(bR[r], sA[r][il], dot);
            const float wv = (float)(codes[e] - 8) * scale + 2.0f * dot; // SCALING = 2
            sT[il][orow] = (__bf16)wv;
        }
    }
    __syncthreads();

    #pragma unroll
    for (int jj = 0; jj < 2; ++jj) {
        const int idx = t + jj * 256;
        const int ir  = idx >> 3;
        const int oc  = (idx & 7) * 8;
        *(bf16x8*)(Wt + (size_t)(i0 + ir) * OUT_F + o0 + oc) =
            *(const bf16x8*)&sT[ir][oc];
    }
}

// ---------------------------------------------------------------------------
// Kernel 2: out[m][n] = sum_k bf16(x[m][k]) * Wt[k][n] + bias[n]
// Block: 256 threads (8 waves) -> 128(M) x 256(N) tile, K-step 32.
// Wave grid 2(M) x 4(N); per-wave register tile 64x64 = 16 WMMA per K-step.
// Double-buffered LDS, single barrier per K-step.
// ---------------------------------------------------------------------------
__global__ __launch_bounds__(256) void qlora_wmma_gemm_kernel(
    const float* __restrict__ x, const __bf16* __restrict__ Wt,
    const float* __restrict__ bias, float* __restrict__ out)
{
    __shared__ __align__(16) __bf16 lA[2][BM * LDA];   // A: [row m][k 0..31]
    __shared__ __align__(16) __bf16 lB[2][32 * LDB];   // B: [row k][n 0..255]

    const int t    = threadIdx.x;
    const int lane = t & 31;
    const int w    = t >> 5;
    const int wm   = w >> 2;         // 0..1 : 64-row half
    const int wn   = w & 3;          // 0..3 : 64-col strip
    const int gm   = blockIdx.y * BM;
    const int gn   = blockIdx.x * BN;
    const int mr   = lane & 15;      // A row within 16x16 tile / C column
    const int hi   = lane >> 4;      // K-half selector for A, M-half for C

    // staging decomposition (per thread)
    const int arow = t >> 3;               // handled rows: arow + 32*j  (A: via idx)
    (void)arow;

    f32x8 acc[4][4];                       // [nt][mt]
    #pragma unroll
    for (int nt = 0; nt < 4; ++nt)
        #pragma unroll
        for (int mt = 0; mt < 4; ++mt) acc[nt][mt] = (f32x8)0.0f;

    // ---- prologue: stage k0 = 0 into buffer 0 ----
    {
        #pragma unroll
        for (int j = 0; j < 4; ++j) {
            const int idx = t + j * 256;
            const int row = idx >> 3;          // 0..127
            const int c4  = (idx & 7) * 4;     // 0..28
            const float4 av = *(const float4*)(x + (size_t)(gm + row) * IN_F + c4);
            bf16x4 p;
            p[0] = (__bf16)av.x; p[1] = (__bf16)av.y;
            p[2] = (__bf16)av.z; p[3] = (__bf16)av.w;
            *(bf16x4*)&lA[0][row * LDA + c4] = p;
        }
        #pragma unroll
        for (int j = 0; j < 4; ++j) {
            const int idx = t + j * 256;
            const int row = idx >> 5;          // 0..31 (= k)
            const int ch  = (idx & 31) * 8;    // 0..248
            *(bf16x8*)&lB[0][row * LDB + ch] =
                *(const bf16x8*)(Wt + (size_t)row * OUT_F + gn + ch);
        }
    }

    for (int k0 = 0; k0 < IN_F; k0 += 32) {
        const int s = (k0 >> 5) & 1;
        __syncthreads();   // stage s visible; previous reads of s^1 finished

        // ---- issue next-stage global loads early ----
        float4 avn[4];
        bf16x8 bvn[4];
        const bool more = (k0 + 32) < IN_F;
        if (more) {
            const int k1 = k0 + 32;
            #pragma unroll
            for (int j = 0; j < 4; ++j) {
                const int idx = t + j * 256;
                const int row = idx >> 3;
                const int c4  = (idx & 7) * 4;
                avn[j] = *(const float4*)(x + (size_t)(gm + row) * IN_F + k1 + c4);
            }
            #pragma unroll
            for (int j = 0; j < 4; ++j) {
                const int idx = t + j * 256;
                const int row = idx >> 5;
                const int ch  = (idx & 31) * 8;
                bvn[j] = *(const bf16x8*)(Wt + (size_t)(k1 + row) * OUT_F + gn + ch);
            }
        }

        // ---- A fragments: lane row = mr; K chunks [8*hi,+8) and [16+8*hi,+8) ----
        union FragU { bf16x16 v; bf16x8 h[2]; };
        FragU af[4];
        #pragma unroll
        for (int mt = 0; mt < 4; ++mt) {
            const __bf16* ab = &lA[s][(wm * 64 + mt * 16 + mr) * LDA];
            af[mt].h[0] = *(const bf16x8*)(ab + hi * 8);
            af[mt].h[1] = *(const bf16x8*)(ab + 16 + hi * 8);
        }
        // ---- B fragments (lane col N = mr; K rows [8*hi,+8) and [16+8*hi,+8)) + WMMA ----
        #pragma unroll
        for (int nt = 0; nt < 4; ++nt) {
            FragU bf;
            const __bf16* bb = &lB[s][wn * 64 + nt * 16 + mr];
            #pragma unroll
            for (int i = 0; i < 8; ++i) {
                bf.v[i]     = bb[(hi * 8 + i) * LDB];
                bf.v[8 + i] = bb[(16 + hi * 8 + i) * LDB];
            }
            #pragma unroll
            for (int mt = 0; mt < 4; ++mt) {
                acc[nt][mt] = __builtin_amdgcn_wmma_f32_16x16x32_bf16(
                    false, af[mt].v, false, bf.v, (short)0, acc[nt][mt], false, false);
            }
        }

        // ---- write next stage (no barrier needed: disjoint buffer) ----
        if (more) {
            #pragma unroll
            for (int j = 0; j < 4; ++j) {
                const int idx = t + j * 256;
                const int row = idx >> 3;
                const int c4  = (idx & 7) * 4;
                bf16x4 p;
                p[0] = (__bf16)avn[j].x; p[1] = (__bf16)avn[j].y;
                p[2] = (__bf16)avn[j].z; p[3] = (__bf16)avn[j].w;
                *(bf16x4*)&lA[s ^ 1][row * LDA + c4] = p;
            }
            #pragma unroll
            for (int j = 0; j < 4; ++j) {
                const int idx = t + j * 256;
                const int row = idx >> 5;
                const int ch  = (idx & 31) * 8;
                *(bf16x8*)&lB[s ^ 1][row * LDB + ch] = bvn[j];
            }
        }
    }

    // ---- epilogue: C/D layout M = r + 8*hi, N = mr ----
    #pragma unroll
    for (int nt = 0; nt < 4; ++nt) {
        const int   col = gn + wn * 64 + nt * 16 + mr;
        const float b   = bias[col];
        #pragma unroll
        for (int mt = 0; mt < 4; ++mt) {
            const int rowbase = gm + wm * 64 + mt * 16 + hi * 8;
            #pragma unroll
            for (int r = 0; r < 8; ++r) {
                out[(size_t)(rowbase + r) * OUT_F + col] = acc[nt][mt][r] + b;
            }
        }
    }
}

// ---------------------------------------------------------------------------
extern "C" void kernel_launch(void* const* d_in, const int* in_sizes, int n_in,
                              void* d_out, int out_size, void* d_ws, size_t ws_size,
                              hipStream_t stream)
{
    (void)in_sizes; (void)n_in; (void)out_size; (void)ws_size;

    const float* x      = (const float*)d_in[0];
    const int*   qcodes = (const int*)  d_in[1];
    const float* scales = (const float*)d_in[2];
    const float* bias   = (const float*)d_in[3];
    const float* loraA  = (const float*)d_in[4];
    const float* loraB  = (const float*)d_in[5];
    float*       out    = (float*)d_out;
    __bf16*      Wt     = (__bf16*)d_ws;   // 4096*4096*2 = 32 MB scratch

    dim3 blk(256);
    // 1) dequant + LoRA fold -> transposed bf16 weight
    dequant_fold_kernel<<<dim3(IN_F / 64, OUT_F / 64), blk, 0, stream>>>(
        qcodes, scales, loraA, loraB, Wt);
    // 2) single fused GEMM + bias (LoRA + dequant folded into Wt)
    qlora_wmma_gemm_kernel<<<dim3(OUT_F / BN, M_TOT / BM), blk, 0, stream>>>(
        x, Wt, bias, out);
}
